// MSARowAttentionWithPairBias_962072674748
// MI455X (gfx1250) — matrix operand order repaired and checked by reading, not compiled, not run on hardware
//
#include <hip/hip_runtime.h>
#include <hip/hip_bf16.h>

typedef _Float16 f16;
typedef __attribute__((ext_vector_type(16))) _Float16 v16h;
typedef __attribute__((ext_vector_type(8)))  _Float16 v8h;
typedef __attribute__((ext_vector_type(8)))  float    v8f;

#define S_  128
#define I_  256
#define CS  256
#define CZ  128
#define H_  8
#define D_  32
#define TOT 256
#define ROWS (S_ * I_)   // 32768

__device__ __forceinline__ v8f wmma16(v16h a, v16h b, v8f c) {
  return __builtin_amdgcn_wmma_f32_16x16x32_f16(false, a, false, b, (short)0, c, false, false);
}

__device__ __forceinline__ v16h mkfrag(v8h lo, v8h hi) {
  v16h r;
#pragma unroll
  for (int j = 0; j < 8; ++j) { r[j] = lo[j]; r[j + 8] = hi[j]; }
  return r;
}

__device__ __forceinline__ float wred_sum32(float v) {
#pragma unroll
  for (int m = 16; m >= 1; m >>= 1) v += __shfl_xor(v, m, 32);
  return v;
}

__device__ __forceinline__ float fast_rcp(float x) {
  return __builtin_amdgcn_rcpf(x);   // single v_rcp_f32 (vs IEEE div sequence)
}

// ---------------------------------------------------------------------------
// Kernel 0: pack f32 weights [K=256, N=256] into f16 WMMA B-fragment order.
// Packed index: ((kt*16 + nt)*32 + lane)*16 + slot ; k = kt*32+(lane/16)*16+slot,
// n = nt*16 + lane%16.  Matrices: 0=Wq 1=Wk 2=Wv 3=Wg 4=Wo.
// ---------------------------------------------------------------------------
__global__ __launch_bounds__(256) void pack_weights(
    const float* __restrict__ Wq, const float* __restrict__ Wk,
    const float* __restrict__ Wv, const float* __restrict__ Wg,
    const float* __restrict__ Wo, f16* __restrict__ dst) {
  int idx  = blockIdx.x * 256 + threadIdx.x;  // 0 .. 5*65536-1
  int mat  = idx >> 16;
  int e    = idx & 65535;
  int slot = e & 15;
  int lane = (e >> 4) & 31;
  int nt   = (e >> 9) & 15;
  int kt   = e >> 13;
  int k = kt * 32 + (lane >> 4) * 16 + slot;
  int n = nt * 16 + (lane & 15);
  const float* W = (mat == 0) ? Wq : (mat == 1) ? Wk : (mat == 2) ? Wv
                   : (mat == 3) ? Wg : Wo;
  dst[idx] = (f16)W[k * 256 + n];
}

// ---------------------------------------------------------------------------
// Kernel 1: fused LayerNorm(msa) + QKVG projections via WMMA.
// One block = 32 rows; LDS holds normalized rows in f16; GEMM over N=1024
// (Wq|Wk|Wv|Wg). Q/K/V scattered to [s][h][i][d] f16, G = sigmoid -> f32.
// ---------------------------------------------------------------------------
__global__ __launch_bounds__(256) void ln_qkvg(
    const float* __restrict__ msa, const float* __restrict__ gln,
    const float* __restrict__ bln, const float* __restrict__ bg,
    const f16* __restrict__ wpk, f16* __restrict__ Qh, f16* __restrict__ Kh,
    f16* __restrict__ Vh, float* __restrict__ Gws) {
  __shared__ _Float16 Asm[32 * 256];
  const int tid  = threadIdx.x;
  const int wave = tid >> 5;
  const int lane = tid & 31;
  const int base = blockIdx.x * 32;

  // ---- LayerNorm: each wave normalizes 4 rows (lane covers 8 cols) ----
#pragma unroll
  for (int r = 0; r < 4; ++r) {
    const int row  = wave * 4 + r;
    const int grow = base + row;
    const float* x = msa + (size_t)grow * CS + lane * 8;
    float v[8];
    float s = 0.f;
#pragma unroll
    for (int j = 0; j < 8; ++j) { v[j] = x[j]; s += v[j]; }
    s = wred_sum32(s);
    const float mu = s * (1.f / 256.f);
    float q = 0.f;
#pragma unroll
    for (int j = 0; j < 8; ++j) { float d = v[j] - mu; q += d * d; }
    q = wred_sum32(q);
    const float rstd = rsqrtf(q * (1.f / 256.f) + 1e-5f);
#pragma unroll
    for (int j = 0; j < 8; ++j) {
      const int c = lane * 8 + j;
      Asm[row * 256 + c] = (f16)((v[j] - mu) * rstd * gln[c] + bln[c]);
    }
  }
  __syncthreads();

  const int hl = lane >> 4;  // half-wave id
  const int ml = lane & 15;

  // ---- GEMM: 2 M-tiles x 64 N-tiles (16 per weight matrix), K=256 ----
  for (int t = wave; t < 128; t += 8) {
    const int mt  = t & 1;
    const int nt  = t >> 1;      // 0..63
    const int mat = nt >> 4;     // 0..3
    const int ntl = nt & 15;
    v8f acc = {};
    const f16* bbase = wpk + (size_t)mat * 65536 + ((size_t)(ntl * 32 + lane) << 4);
#pragma unroll
    for (int kt = 0; kt < 8; ++kt) {
      const _Float16* ar = &Asm[(mt * 16 + ml) * 256 + kt * 32 + hl * 16];
      v16h a = mkfrag(*(const v8h*)ar, *(const v8h*)(ar + 8));
      const f16* bp = bbase + (size_t)kt * (16 * 32 * 16);
      if (kt < 7) __builtin_prefetch(bp + 16 * 32 * 16, 0, 0);  // global_prefetch_b8
      v16h b = *(const v16h*)bp;
      acc = wmma16(a, b, acc);
    }
#pragma unroll
    for (int v = 0; v < 8; ++v) {
      const int m    = v + hl * 8;
      const int grow = base + mt * 16 + m;
      const int s    = grow >> 8;
      const int i    = grow & 255;
      const int c    = ntl * 16 + ml;
      const float val = acc[v];
      if (mat < 3) {
        const int h = c >> 5, d = c & 31;
        const size_t off = ((((size_t)s * H_ + h) * I_ + i) * D_ + d);
        f16* dstp = (mat == 0) ? Qh : (mat == 1) ? Kh : Vh;
        dstp[off] = (f16)val;
      } else {
        Gws[(size_t)grow * TOT + c] = fast_rcp(1.f + __expf(-(val + bg[c])));
      }
    }
  }
}

// ---------------------------------------------------------------------------
// Kernel 2: pair LayerNorm + Wp projection -> bias[h][i][j] (f32).
// One wave per (i,j); lane covers 4 of the 128 channels.
// ---------------------------------------------------------------------------
__global__ __launch_bounds__(256) void pair_bias(
    const float* __restrict__ pair, const float* __restrict__ gln,
    const float* __restrict__ bln, const float* __restrict__ Wp,
    float* __restrict__ bias) {
  const int lane = threadIdx.x & 31;
  const int wid  = blockIdx.x * 8 + (threadIdx.x >> 5);  // 0..65535
  const int i = wid >> 8;
  const int j = wid & 255;
  const float* x = pair + ((size_t)i * I_ + j) * CZ + lane * 4;
  float v[4];
  float s = 0.f;
#pragma unroll
  for (int c = 0; c < 4; ++c) { v[c] = x[c]; s += v[c]; }
  s = wred_sum32(s);
  const float mu = s * (1.f / 128.f);
  float q = 0.f;
#pragma unroll
  for (int c = 0; c < 4; ++c) { float d = v[c] - mu; q += d * d; }
  q = wred_sum32(q);
  const float rstd = rsqrtf(q * (1.f / 128.f) + 1e-5f);
  float z[4];
#pragma unroll
  for (int c = 0; c < 4; ++c) {
    const int cc = lane * 4 + c;
    z[c] = (v[c] - mu) * rstd * gln[cc] + bln[cc];
  }
#pragma unroll
  for (int h = 0; h < H_; ++h) {
    float p = 0.f;
#pragma unroll
    for (int c = 0; c < 4; ++c) p += z[c] * Wp[(lane * 4 + c) * H_ + h];
    p = wred_sum32(p);
    if (lane == 0) bias[((size_t)h * I_ + i) * I_ + j] = p;
  }
}

// ---------------------------------------------------------------------------
// Kernel 3: attention for one (s,h) per block.
// K tile staged to LDS via async-to-LDS DMA (ASYNCcnt); QK^T (D=32 -> 1 WMMA
// per 16x16 tile), +bias, softmax in-register with half-wave shuffles,
// P staged f16 in LDS, P@V via WMMA (V transposed in LDS), gated output.
// ---------------------------------------------------------------------------
__global__ __launch_bounds__(256) void attn_kernel(
    const f16* __restrict__ Qh, const f16* __restrict__ Kh,
    const f16* __restrict__ Vh, const float* __restrict__ bias,
    const float* __restrict__ Gws, f16* __restrict__ att) {
  extern __shared__ __align__(64) _Float16 smem[];
  _Float16* vt = smem;                       // [32 d][256 j]   (V transposed)
  _Float16* kb = smem + 32 * 256;            // [256 j][32 d]   (K row-major)
  _Float16* pb = smem + 32 * 256 + 256 * 32; // 8 waves * [16][256]

  const int tid  = threadIdx.x;
  const int wave = tid >> 5;
  const int lane = tid & 31;
  const int sh = blockIdx.x;
  const int s  = sh >> 3;
  const int h  = sh & 7;
  const size_t headoff = (((size_t)s * H_ + h) * I_) * D_;
  const f16* Qb = Qh + headoff;
  const f16* Kb = Kh + headoff;
  const f16* Vb = Vh + headoff;

  // --- async DMA: stage K tile (16 KB) into LDS, 16B chunks, 4 per thread ---
  {
    const unsigned kb_lds = (unsigned)(uintptr_t)kb;   // low 32b = LDS offset
    const unsigned long long kgl = (unsigned long long)(uintptr_t)Kb;
#pragma unroll
    for (int c = 0; c < 4; ++c) {
      const int chunk = tid + c * 256;                 // 0..1023
      unsigned lv = kb_lds + chunk * 16;
      unsigned long long ga = kgl + (unsigned long long)chunk * 16;
      asm volatile("global_load_async_to_lds_b128 %0, %1, off"
                   :: "v"(lv), "v"(ga) : "memory");
    }
  }
  // stage V transposed: vt[d][j] = V[j][d]
  for (int e = tid; e < I_ * D_; e += 256) {
    const int j = e >> 5, d = e & 31;
    vt[d * 256 + j] = Vb[e];
  }
  asm volatile("s_wait_asynccnt 0x0" ::: "memory");
  __syncthreads();

  const int hl = lane >> 4;
  const int ml = lane & 15;
  const float scale = 0.17677669529663687f;  // 1/sqrt(32)
  _Float16* pw = pb + wave * 16 * 256;

  for (int it = wave; it < 16; it += 8) {  // uniform: every wave does 2 tiles
    // Q A-fragment (single K-step since D=32)
    const f16* qr = Qb + (it * 16 + ml) * D_ + hl * 16;
    v16h aq = mkfrag(*(const v8h*)qr, *(const v8h*)(qr + 8));

    v8f sc[16];
#pragma unroll
    for (int jt = 0; jt < 16; ++jt) {
      const _Float16* kr = kb + (jt * 16 + ml) * D_ + hl * 16;
      v16h bk = *(const v16h*)kr;
      v8f z = {};
      sc[jt] = wmma16(aq, bk, z);
    }
    // scale + pair bias  (C frag: row = v + 8*hl, col = jt*16 + ml)
    const float* bh = bias + (size_t)h * I_ * I_;
#pragma unroll
    for (int jt = 0; jt < 16; ++jt)
#pragma unroll
      for (int v = 0; v < 8; ++v) {
        const int i = it * 16 + v + hl * 8;
        const int j = jt * 16 + ml;
        sc[jt][v] = sc[jt][v] * scale + bh[(size_t)i * I_ + j];
      }
    // softmax over j: reduce across the 16 lanes of each half-wave
#pragma unroll
    for (int v = 0; v < 8; ++v) {
      float mx = -1e30f;
#pragma unroll
      for (int jt = 0; jt < 16; ++jt) mx = fmaxf(mx, sc[jt][v]);
#pragma unroll
      for (int m = 8; m >= 1; m >>= 1) mx = fmaxf(mx, __shfl_xor(mx, m, 32));
      float sum = 0.f;
#pragma unroll
      for (int jt = 0; jt < 16; ++jt) {
        const float e = __expf(sc[jt][v] - mx);
        sc[jt][v] = e;
        sum += e;
      }
#pragma unroll
      for (int m = 8; m >= 1; m >>= 1) sum += __shfl_xor(sum, m, 32);
      const float inv = fast_rcp(sum);
#pragma unroll
      for (int jt = 0; jt < 16; ++jt) sc[jt][v] *= inv;
    }
    // stage P (f16) into this wave's LDS region, row-major [16][256]
#pragma unroll
    for (int jt = 0; jt < 16; ++jt)
#pragma unroll
      for (int v = 0; v < 8; ++v) {
        const int m = v + hl * 8;
        const int j = jt * 16 + ml;
        pw[m * 256 + j] = (f16)sc[jt][v];
      }
    // P @ V : out tile [16 x 32], K = 256 (8 WMMA steps)
#pragma unroll
    for (int nt = 0; nt < 2; ++nt) {
      v8f o = {};
#pragma unroll
      for (int kt = 0; kt < 8; ++kt) {
        const _Float16* pr = pw + ml * 256 + kt * 32 + hl * 16;
        v16h ap = mkfrag(*(const v8h*)pr, *(const v8h*)(pr + 8));
        const _Float16* vr = vt + (nt * 16 + ml) * 256 + kt * 32 + hl * 16;
        v16h bv = *(const v16h*)vr;
        o = wmma16(ap, bv, o);
      }
#pragma unroll
      for (int v = 0; v < 8; ++v) {
        const int i    = it * 16 + v + hl * 8;
        const int d    = nt * 16 + ml;
        const int grow = s * I_ + i;
        const int c    = h * D_ + d;
        const float g  = Gws[(size_t)grow * TOT + c];
        att[(size_t)grow * TOT + c] = (f16)(o[v] * g);
      }
    }
  }
}

// ---------------------------------------------------------------------------
// Kernel 4: output projection  out = att(f16) @ Wo + bo   (WMMA, A from global)
// ---------------------------------------------------------------------------
__global__ __launch_bounds__(256) void out_proj(
    const f16* __restrict__ att, const f16* __restrict__ wo_pk,
    const float* __restrict__ bo, float* __restrict__ out) {
  const int tid  = threadIdx.x;
  const int wave = tid >> 5;
  const int lane = tid & 31;
  const int base = blockIdx.x * 32;
  const int hl = lane >> 4;
  const int ml = lane & 15;

  for (int t = wave; t < 32; t += 8) {
    const int mt = t & 1;
    const int nt = t >> 1;  // 0..15
    v8f acc = {};
    const f16* bbase = wo_pk + ((size_t)(nt * 32 + lane) << 4);
#pragma unroll
    for (int kt = 0; kt < 8; ++kt) {
      const f16* ar = att + (size_t)(base + mt * 16 + ml) * TOT + kt * 32 + hl * 16;
      v16h a = mkfrag(*(const v8h*)ar, *(const v8h*)(ar + 8));
      const f16* bp = bbase + (size_t)kt * (16 * 32 * 16);
      if (kt < 7) __builtin_prefetch(bp + 16 * 32 * 16, 0, 0);
      v16h b = *(const v16h*)bp;
      acc = wmma16(a, b, acc);
    }
#pragma unroll
    for (int v = 0; v < 8; ++v) {
      const int row = base + mt * 16 + v + hl * 8;
      const int n   = nt * 16 + ml;
      out[(size_t)row * CS + n] = acc[v] + bo[n];
    }
  }
}

// ---------------------------------------------------------------------------
extern "C" void kernel_launch(void* const* d_in, const int* in_sizes, int n_in,
                              void* d_out, int out_size, void* d_ws,
                              size_t ws_size, hipStream_t stream) {
  (void)in_sizes; (void)n_in; (void)out_size; (void)ws_size;
  const float* msa   = (const float*)d_in[0];
  const float* pair  = (const float*)d_in[1];
  const float* ln_mg = (const float*)d_in[2];
  const float* ln_mb = (const float*)d_in[3];
  const float* ln_pg = (const float*)d_in[4];
  const float* ln_pb = (const float*)d_in[5];
  const float* Wq    = (const float*)d_in[6];
  const float* Wk    = (const float*)d_in[7];
  const float* Wv    = (const float*)d_in[8];
  const float* Wp    = (const float*)d_in[9];
  const float* Wg    = (const float*)d_in[10];
  const float* bg    = (const float*)d_in[11];
  const float* Wo    = (const float*)d_in[12];
  const float* bo    = (const float*)d_in[13];
  float* out = (float*)d_out;

  char* ws = (char*)d_ws;
  const size_t szWpk = (size_t)5 * 65536 * sizeof(f16);   // 655360
  const size_t szQKV = (size_t)ROWS * TOT * sizeof(f16);  // 16 MB each
  const size_t szG   = (size_t)ROWS * TOT * sizeof(float);
  const size_t szB   = (size_t)H_ * I_ * I_ * sizeof(float);
  f16*   wpk  = (f16*)(ws);
  f16*   Qh   = (f16*)(ws + szWpk);
  f16*   Kh   = (f16*)(ws + szWpk + szQKV);
  f16*   Vh   = (f16*)(ws + szWpk + 2 * szQKV);
  float* Gws  = (float*)(ws + szWpk + 3 * szQKV);
  float* bias = (float*)(ws + szWpk + 3 * szQKV + szG);
  f16*   attb = (f16*)(ws + szWpk + 3 * szQKV + szG + szB);

  pack_weights<<<1280, 256, 0, stream>>>(Wq, Wk, Wv, Wg, Wo, wpk);
  ln_qkvg<<<ROWS / 32, 256, 0, stream>>>(msa, ln_mg, ln_mb, bg, wpk, Qh, Kh,
                                         Vh, Gws);
  pair_bias<<<(I_ * I_) / 8, 256, 0, stream>>>(pair, ln_pg, ln_pb, Wp, bias);
  const size_t lds3 =
      (size_t)(32 * 256 + 256 * 32 + 8 * 16 * 256) * sizeof(_Float16); // 96 KB
  attn_kernel<<<S_ * H_, 256, lds3, stream>>>(Qh, Kh, Vh, bias, Gws, attb);
  out_proj<<<ROWS / 32, 256, 0, stream>>>(attb, wpk + (size_t)4 * 65536, bo,
                                          out);
}
